// TopoGCN_v3_skip_59828894433566
// MI455X (gfx1250) — matrix operand order repaired and checked by reading, not compiled, not run on hardware
//
#include <hip/hip_runtime.h>
#include <hip/hip_bf16.h>

// ---------------------------------------------------------------------------
// TopoGCN_v3 on MI455X (gfx1250, wave32, WMMA).
// GEMMs: BF16 WMMA (v_wmma_f32_16x16x32_bf16, f32 accumulate), fragments
// fetched as contiguous b128 loads per the CDNA5 16-bit A/B VGPR layouts.
// Edge scatter: one wave32 per edge, float4 gathers + global f32 atomics.
// ---------------------------------------------------------------------------

typedef __bf16 bf16_t;
typedef bf16_t v16bf __attribute__((ext_vector_type(16)));
typedef float  v8f   __attribute__((ext_vector_type(8)));

#define NNODES 20000
#define HID    512
#define KBIG   1029     // FEAT_DIM / GAT_IN
#define KPAD   1056     // 33 * 32
#define XLD    1056     // row stride of packed bf16 activation buffer

union FragAB { uint4 q[2]; unsigned int u[8]; v16bf v; };
union FragC  { float f[8]; v8f v; };

__device__ __forceinline__ bf16_t f2bf(float x) {
    union { float f; unsigned int u; } in; in.f = x;
    unsigned int r = in.u + 0x7FFFu + ((in.u >> 16) & 1u);   // RNE
    union { unsigned short s; bf16_t b; } out;
    out.s = (unsigned short)(r >> 16);
    return out.b;
}

// ---------------------------------------------------------------------------
// Build Xb[N,1056] bf16: cols [0,512)=feat, [512,1024)=goal, [1024,1029)=info,
// [1029,1056)=0.  Cols >=512 stay constant for the whole forward pass.
// ---------------------------------------------------------------------------
__global__ __launch_bounds__(256) void k_pack_input(
    const float* __restrict__ feat, const float* __restrict__ goal,
    const float* __restrict__ info, bf16_t* __restrict__ Xb, int nnodes)
{
    long idx = (long)blockIdx.x * blockDim.x + threadIdx.x;
    long total = (long)nnodes * XLD;
    if (idx >= total) return;
    int row = (int)(idx / XLD);
    int col = (int)(idx % XLD);
    float v;
    if      (col < 512)  v = feat[(long)row * 512 + col];
    else if (col < 1024) v = goal[(long)row * 512 + (col - 512)];
    else if (col < 1029) v = info[(long)row * 5 + (col - 1024)];
    else                 v = 0.f;
    Xb[idx] = f2bf(v);
}

// W[K,512] f32 (in-dim major) -> WT[512][Kpad] bf16, K zero-padded
__global__ __launch_bounds__(256) void k_convert_wT(
    const float* __restrict__ W, bf16_t* __restrict__ WT, int K, int Kpad)
{
    long idx = (long)blockIdx.x * blockDim.x + threadIdx.x;
    long total = (long)512 * Kpad;
    if (idx >= total) return;
    int n = (int)(idx / Kpad);
    int k = (int)(idx % Kpad);
    WT[idx] = f2bf(k < K ? W[(long)k * 512 + n] : 0.f);
}

// H f32 [N,512] -> bf16 written into dst with row stride dstLd (Xb cols 0..511)
__global__ __launch_bounds__(256) void k_f32_to_bf16(
    const float* __restrict__ src, bf16_t* __restrict__ dst, int dstLd, int nnodes)
{
    long idx = (long)blockIdx.x * blockDim.x + threadIdx.x;
    if (idx >= (long)nnodes * 512) return;
    int row = (int)(idx >> 9);
    int col = (int)(idx & 511);
    dst[(long)row * dstLd + col] = f2bf(src[idx]);
}

// ---------------------------------------------------------------------------
// C[N,512] = act(A_bf16[N,lda] @ WT_bf16[512][ldb] + bias)
// grid.x = N/32; 8 wave32 per block. Each wave: 2 row tiles x 4 col tiles
// (cols [64w, 64w+64)), 8 WMMA per K-step fed by 12 b128 loads.
// A frag per lane: two b128 at byte offs base+2k0 (+32) where
//   base = ((row)*lda + 8*half)*2            (K = k0+8*half+0..7 | +16..)
// B frag per lane: two b128 at ((n*ldb + 8*half)*2 + 2*k0) (+32)
//   (K = k0+8*half+0..7 | +16..)
// ---------------------------------------------------------------------------
__global__ __launch_bounds__(256) void k_gemm_bf16_wmma(
    const bf16_t* __restrict__ A, int lda,
    const bf16_t* __restrict__ BT, int ldb,
    const float* __restrict__ bias,
    float* __restrict__ C, int Kpad, int act)
{
    const int lane = threadIdx.x & 31;
    const int wave = threadIdx.x >> 5;          // 0..7
    const int row0 = blockIdx.x << 5;           // 32 rows per block
    const int m    = lane & 15;
    const int half = lane >> 4;

    const char* Ab = reinterpret_cast<const char*>(A);
    const char* Bb = reinterpret_cast<const char*>(BT);

    // 32-bit byte offsets (buffers are < 64 MB)
    unsigned int abase[2];
#pragma unroll
    for (int r = 0; r < 2; ++r)
        abase[r] = (unsigned int)(((row0 + (r << 4) + m) * lda + (half << 3)) << 1);

    unsigned int bbase[4];
#pragma unroll
    for (int t = 0; t < 4; ++t) {
        int n = (wave << 6) + (t << 4) + m;
        bbase[t] = (unsigned int)((n * ldb + (half << 3)) << 1);
    }

    FragC acc[2][4];
#pragma unroll
    for (int r = 0; r < 2; ++r)
#pragma unroll
        for (int t = 0; t < 4; ++t)
#pragma unroll
            for (int j = 0; j < 8; ++j) acc[r][t].f[j] = 0.f;

    for (int k0 = 0; k0 < Kpad; k0 += 32) {
        const unsigned int kb = (unsigned int)(k0 << 1);  // byte advance
        FragAB a0, a1;
        a0.q[0] = *reinterpret_cast<const uint4*>(Ab + abase[0] + kb);
        a0.q[1] = *reinterpret_cast<const uint4*>(Ab + abase[0] + kb + 32);
        a1.q[0] = *reinterpret_cast<const uint4*>(Ab + abase[1] + kb);
        a1.q[1] = *reinterpret_cast<const uint4*>(Ab + abase[1] + kb + 32);
#pragma unroll
        for (int t = 0; t < 4; ++t) {
            FragAB b;
            b.q[0] = *reinterpret_cast<const uint4*>(Bb + bbase[t] + kb);
            b.q[1] = *reinterpret_cast<const uint4*>(Bb + bbase[t] + kb + 32);
            acc[0][t].v = __builtin_amdgcn_wmma_f32_16x16x32_bf16(
                false, a0.v, false, b.v, (short)0, acc[0][t].v, false, false);
            acc[1][t].v = __builtin_amdgcn_wmma_f32_16x16x32_bf16(
                false, a1.v, false, b.v, (short)0, acc[1][t].v, false, false);
        }
    }

    // C layout: VGPR j -> row M = j + 8*half, col N = m
#pragma unroll
    for (int r = 0; r < 2; ++r)
#pragma unroll
        for (int t = 0; t < 4; ++t) {
            int n = (wave << 6) + (t << 4) + m;
            float bv = bias ? bias[n] : 0.f;
#pragma unroll
            for (int j = 0; j < 8; ++j) {
                int mm = (r << 4) + j + (half << 3);
                float v = acc[r][t].f[j] + bv;
                if (act) v = v > 0.f ? v : 0.f;       // relu
                C[(long)(row0 + mm) * 512 + n] = v;
            }
        }
}

// ---------------------------------------------------------------------------
// s1[n] = H[n]·a[:512], s2[n] = H[n]·a[512:]; one wave per node
// ---------------------------------------------------------------------------
__global__ __launch_bounds__(256) void k_node_dots(
    const float* __restrict__ H, const float* __restrict__ avec,
    float* __restrict__ s1, float* __restrict__ s2, int nnodes)
{
    int w    = (int)(((long)blockIdx.x * blockDim.x + threadIdx.x) >> 5);
    int lane = threadIdx.x & 31;
    if (w >= nnodes) return;
    const float* h = H + (long)w * 512;
    float d1 = 0.f, d2 = 0.f;
#pragma unroll
    for (int i = 0; i < 16; ++i) {
        int c = lane + (i << 5);
        float hv = h[c];
        d1 += hv * avec[c];
        d2 += hv * avec[512 + c];
    }
#pragma unroll
    for (int off = 16; off > 0; off >>= 1) {
        d1 += __shfl_xor(d1, off, 32);
        d2 += __shfl_xor(d2, off, 32);
    }
    if (lane == 0) { s1[w] = d1; s2[w] = d2; }
}

// ---------------------------------------------------------------------------
// One wave per edge: w = exp(-leaky_relu(s1[src]+s2[dst])),
// rowsum[src] += w ; acc[src,:] += w * H[dst,:]   (f32 global atomics)
// ---------------------------------------------------------------------------
__global__ __launch_bounds__(256) void k_edge_scatter(
    const int* __restrict__ esrc, const int* __restrict__ edst,
    const float* __restrict__ s1, const float* __restrict__ s2,
    const float* __restrict__ H, float* __restrict__ acc,
    float* __restrict__ rowsum, int nedges)
{
    int e    = (int)(((long)blockIdx.x * blockDim.x + threadIdx.x) >> 5);
    int lane = threadIdx.x & 31;
    if (e >= nedges) return;
    int s = esrc[e], d = edst[e];
    float logit = s1[s] + s2[d];
    float lr = logit > 0.f ? logit : 0.2f * logit;
    float w  = __expf(-lr);
    if (lane == 0)
        __hip_atomic_fetch_add(rowsum + s, w, __ATOMIC_RELAXED, __HIP_MEMORY_SCOPE_AGENT);
    const float4* hv = reinterpret_cast<const float4*>(H + (long)d * 512);
    float* out = acc + (long)s * 512;
#pragma unroll
    for (int it = 0; it < 4; ++it) {
        int c4 = (it << 5) + lane;     // float4 index; 128 floats per iter across wave
        float4 v = hv[c4];
        int c = c4 << 2;
        __hip_atomic_fetch_add(out + c + 0, w * v.x, __ATOMIC_RELAXED, __HIP_MEMORY_SCOPE_AGENT);
        __hip_atomic_fetch_add(out + c + 1, w * v.y, __ATOMIC_RELAXED, __HIP_MEMORY_SCOPE_AGENT);
        __hip_atomic_fetch_add(out + c + 2, w * v.z, __ATOMIC_RELAXED, __HIP_MEMORY_SCOPE_AGENT);
        __hip_atomic_fetch_add(out + c + 3, w * v.w, __ATOMIC_RELAXED, __HIP_MEMORY_SCOPE_AGENT);
    }
}

// h' = acc/(rowsum+1e-16); x = relu(elu(h')); write bf16 into Xb cols [0,512)
__global__ __launch_bounds__(256) void k_gat_finish(
    const float* __restrict__ acc, const float* __restrict__ rowsum,
    bf16_t* __restrict__ Xb, int nnodes)
{
    long idx = (long)blockIdx.x * blockDim.x + threadIdx.x;
    if (idx >= (long)nnodes * 512) return;
    int row = (int)(idx >> 9);
    int col = (int)(idx & 511);
    float v = acc[idx] / (rowsum[row] + 1e-16f);
    v = v > 0.f ? v : (__expf(v) - 1.f);   // elu
    v = v > 0.f ? v : 0.f;                 // relu
    Xb[(long)row * XLD + col] = f2bf(v);
}

// out[n] = sigmoid(H2[n]·w2 + b2); one wave per node
__global__ __launch_bounds__(256) void k_val_head(
    const float* __restrict__ H2, const float* __restrict__ w2,
    const float* __restrict__ b2, float* __restrict__ out, int nnodes)
{
    int w    = (int)(((long)blockIdx.x * blockDim.x + threadIdx.x) >> 5);
    int lane = threadIdx.x & 31;
    if (w >= nnodes) return;
    const float* h = H2 + (long)w * 512;
    float d = 0.f;
#pragma unroll
    for (int i = 0; i < 16; ++i) { int c = lane + (i << 5); d += h[c] * w2[c]; }
#pragma unroll
    for (int off = 16; off > 0; off >>= 1) d += __shfl_xor(d, off, 32);
    if (lane == 0) out[w] = 1.f / (1.f + __expf(-(d + b2[0])));
}

// ---------------------------------------------------------------------------
extern "C" void kernel_launch(void* const* d_in, const int* in_sizes, int n_in,
                              void* d_out, int out_size, void* d_ws, size_t ws_size,
                              hipStream_t stream) {
    const float* feat  = (const float*)d_in[0];
    const float* goal  = (const float*)d_in[1];
    const float* info  = (const float*)d_in[2];
    const int*   esrc  = (const int*)d_in[3];
    const int*   edst  = (const int*)d_in[4];
    const float* enc_w[3] = { (const float*)d_in[5], (const float*)d_in[7], (const float*)d_in[9] };
    const float* enc_b[3] = { (const float*)d_in[6], (const float*)d_in[8], (const float*)d_in[10] };
    const float* val_w[3] = { (const float*)d_in[11], (const float*)d_in[13], (const float*)d_in[15] };
    const float* val_b[3] = { (const float*)d_in[12], (const float*)d_in[14], (const float*)d_in[16] };
    const float* gat_W[5]; const float* gat_a[5];
    for (int i = 0; i < 5; ++i) { gat_W[i] = (const float*)d_in[17 + 2*i]; gat_a[i] = (const float*)d_in[18 + 2*i]; }

    const int N = in_sizes[0] / 512;
    const int E = in_sizes[3];
    float* out = (float*)d_out;

    // ---- workspace carve-out (256B aligned) ----
    char* ws = (char*)d_ws;
    size_t off = 0;
    auto carve = [&](size_t bytes) -> void* {
        void* p = ws + off; off = (off + bytes + 255) & ~(size_t)255; return p;
    };
    bf16_t* Xb     = (bf16_t*)carve((size_t)N * XLD * sizeof(bf16_t));  // 42.2 MB
    float*  H      = (float*) carve((size_t)N * 512 * sizeof(float));   // 41.0 MB
    float*  H2     = (float*) carve((size_t)N * 512 * sizeof(float));   // 41.0 MB
    float*  s1     = (float*) carve((size_t)N * sizeof(float));
    float*  s2     = (float*) carve((size_t)N * sizeof(float));
    float*  rowsum = (float*) carve((size_t)N * sizeof(float));
    bf16_t* WT     = (bf16_t*)carve((size_t)512 * KPAD * sizeof(bf16_t)); // reused per GEMM
    (void)ws_size; (void)n_in; (void)out_size;

    const int TPB = 256;
    const int gemmGrid = N / 32;                         // 625; N divides 32
    const int gPack  = (int)(((long)N * XLD + TPB - 1) / TPB);
    const int gCvtB  = (int)(((long)512 * KPAD + TPB - 1) / TPB);
    const int gCvtS  = (int)(((long)512 * 512 + TPB - 1) / TPB);
    const int gElem  = (int)(((long)N * 512 + TPB - 1) / TPB);
    const int gWaveN = (int)(((long)N * 32 + TPB - 1) / TPB);
    const int gWaveE = (int)(((long)E * 32 + TPB - 1) / TPB);

    k_pack_input<<<gPack, TPB, 0, stream>>>(feat, goal, info, Xb, N);

    // ---- encoder MLP: 1029 -> 512 -> 512 -> 512 ----
    k_convert_wT<<<gCvtB, TPB, 0, stream>>>(enc_w[0], WT, KBIG, KPAD);
    k_gemm_bf16_wmma<<<gemmGrid, TPB, 0, stream>>>(Xb, XLD, WT, KPAD, enc_b[0], H, KPAD, 1);
    k_f32_to_bf16<<<gElem, TPB, 0, stream>>>(H, Xb, XLD, N);

    k_convert_wT<<<gCvtS, TPB, 0, stream>>>(enc_w[1], WT, 512, 512);
    k_gemm_bf16_wmma<<<gemmGrid, TPB, 0, stream>>>(Xb, XLD, WT, 512, enc_b[1], H, 512, 1);
    k_f32_to_bf16<<<gElem, TPB, 0, stream>>>(H, Xb, XLD, N);

    k_convert_wT<<<gCvtS, TPB, 0, stream>>>(enc_w[2], WT, 512, 512);
    k_gemm_bf16_wmma<<<gemmGrid, TPB, 0, stream>>>(Xb, XLD, WT, 512, enc_b[2], H, 512, 0);
    k_f32_to_bf16<<<gElem, TPB, 0, stream>>>(H, Xb, XLD, N);

    // ---- 5 GAT layers ----
    for (int i = 0; i < 5; ++i) {
        k_convert_wT<<<gCvtB, TPB, 0, stream>>>(gat_W[i], WT, KBIG, KPAD);
        k_gemm_bf16_wmma<<<gemmGrid, TPB, 0, stream>>>(Xb, XLD, WT, KPAD, nullptr, H, KPAD, 0);
        k_node_dots<<<gWaveN, TPB, 0, stream>>>(H, gat_a[i], s1, s2, N);
        hipMemsetAsync(H2, 0, (size_t)N * 512 * sizeof(float), stream);
        hipMemsetAsync(rowsum, 0, (size_t)N * sizeof(float), stream);
        k_edge_scatter<<<gWaveE, TPB, 0, stream>>>(esrc, edst, s1, s2, H, H2, rowsum, E);
        k_gat_finish<<<gElem, TPB, 0, stream>>>(H2, rowsum, Xb, N);
    }

    // ---- value head: 1029 -> 512 -> 512 -> 1, sigmoid ----
    k_convert_wT<<<gCvtB, TPB, 0, stream>>>(val_w[0], WT, KBIG, KPAD);
    k_gemm_bf16_wmma<<<gemmGrid, TPB, 0, stream>>>(Xb, XLD, WT, KPAD, val_b[0], H, KPAD, 1);
    k_f32_to_bf16<<<gElem, TPB, 0, stream>>>(H, Xb, XLD, N);

    k_convert_wT<<<gCvtS, TPB, 0, stream>>>(val_w[1], WT, 512, 512);
    k_gemm_bf16_wmma<<<gemmGrid, TPB, 0, stream>>>(Xb, XLD, WT, 512, val_b[1], H2, 512, 1);

    k_val_head<<<gWaveN, TPB, 0, stream>>>(H2, val_w[2], val_b[2], out, N);
}
